// NetworkFull_9758165696599
// MI455X (gfx1250) — matrix operand order repaired and checked by reading, not compiled, not run on hardware
//
#include <hip/hip_runtime.h>
#include <math.h>

// ---------------------------------------------------------------------------
// Types
// ---------------------------------------------------------------------------
typedef _Float16 h16;
typedef __attribute__((ext_vector_type(16))) _Float16 v16h;
typedef __attribute__((ext_vector_type(8)))  _Float16 v8h;
typedef __attribute__((ext_vector_type(8)))  float    v8f;
typedef __attribute__((ext_vector_type(4)))  unsigned int u32x4;
typedef __attribute__((ext_vector_type(8)))  int          i32x8;
typedef __attribute__((ext_vector_type(4)))  int          i32x4;

#define BATCH 16
#define NPTS  2048
#define CAGEP 42

// ---------------------------------------------------------------------------
// Weight conversion: f32 [O][K] -> f16 [Opad][Kp], zero padded
// ---------------------------------------------------------------------------
__global__ void convert_w(const float* __restrict__ src, h16* __restrict__ dst,
                          int O, int K, int Opad, int Kp) {
  int i = blockIdx.x * blockDim.x + threadIdx.x;
  if (i >= Opad * Kp) return;
  int o = i / Kp, k = i % Kp;
  dst[i] = (o < O && k < K) ? (h16)src[(size_t)o * K + k] : (h16)0.f;
}

// pc (B,3,N) -> xyz (B,N,3)
__global__ void transpose_pc(const float* __restrict__ pc, float* __restrict__ xyz) {
  int i = blockIdx.x * blockDim.x + threadIdx.x;
  if (i >= BATCH * NPTS * 3) return;
  int c = i % 3, n = (i / 3) % NPTS, b = i / (3 * NPTS);
  xyz[i] = pc[((size_t)b * 3 + c) * NPTS + n];
}

// ---------------------------------------------------------------------------
// Farthest point sampling: one block per batch, serial over S, parallel over N
// ---------------------------------------------------------------------------
__global__ __launch_bounds__(256) void fps_kernel(const float* __restrict__ xyz,
                                                  int* __restrict__ idx, int N, int S) {
  __shared__ float dist[NPTS];
  __shared__ float rmx[256];
  __shared__ int   rid[256];
  int b = blockIdx.x, t = threadIdx.x;
  const float* xb = xyz + (size_t)b * N * 3;
  for (int n = t; n < N; n += 256) dist[n] = 1e10f;
  __syncthreads();
  int far = 0;
  for (int i = 0; i < S; ++i) {
    if (t == 0) idx[(size_t)b * S + i] = far;
    float cx = xb[far * 3 + 0], cy = xb[far * 3 + 1], cz = xb[far * 3 + 2];
    float lm = -1.0f; int la = 0;
    for (int n = t; n < N; n += 256) {
      float dx = xb[n * 3 + 0] - cx, dy = xb[n * 3 + 1] - cy, dz = xb[n * 3 + 2] - cz;
      float d = dx * dx + dy * dy + dz * dz;
      float nd = fminf(dist[n], d);
      dist[n] = nd;
      if (nd > lm) { lm = nd; la = n; }
    }
    rmx[t] = lm; rid[t] = la;
    __syncthreads();
    for (int off = 128; off > 0; off >>= 1) {
      if (t < off && rmx[t + off] > rmx[t]) { rmx[t] = rmx[t + off]; rid[t] = rid[t + off]; }
      __syncthreads();
    }
    far = rid[0];
    __syncthreads();
  }
}

// nxyz[b][s][c] = xyz[b][idx[b][s]][c]
__global__ void gather_xyz(const float* __restrict__ xyz, const int* __restrict__ idx,
                           float* __restrict__ nxyz, int N, int S) {
  int i = blockIdx.x * blockDim.x + threadIdx.x;
  if (i >= BATCH * S * 3) return;
  int c = i % 3, s = (i / 3) % S, b = i / (3 * S);
  nxyz[i] = xyz[((size_t)b * N + idx[(size_t)b * S + s]) * 3 + c];
}

// ---------------------------------------------------------------------------
// Ball query + grouping (wave32 ballot scan): writes GEMM RHS X[b][s*32+j][Kp]
// channels: [0,3) rel xyz, [3,3+C) gathered prev features, rest 0
// ---------------------------------------------------------------------------
__global__ __launch_bounds__(32) void ballgroup(const float* __restrict__ xyz,
                                                const float* __restrict__ nxyz,
                                                const h16* __restrict__ fprev,
                                                h16* __restrict__ X,
                                                int N, int S, int C, int Cp, int Kp, float r2) {
  __shared__ int gidx[32];
  __shared__ int scnt;
  int lane = threadIdx.x;
  int b = blockIdx.y, s = blockIdx.x;
  const float* q = nxyz + ((size_t)b * S + s) * 3;
  float qx = q[0], qy = q[1], qz = q[2];
  int cnt = 0;
  for (int base = 0; base < N && cnt < 32; base += 32) {
    int n = base + lane;
    bool inside = false;
    if (n < N) {
      const float* p = xyz + ((size_t)b * N + n) * 3;
      float dx = p[0] - qx, dy = p[1] - qy, dz = p[2] - qz;
      inside = (dx * dx + dy * dy + dz * dz < r2);
    }
    unsigned m32 = (unsigned)__ballot(inside);   // wave32: 32 meaningful bits
    if (lane == 0) {
      while (m32 && cnt < 32) {
        int bit = __builtin_ctz(m32);
        gidx[cnt++] = base + bit;
        m32 &= m32 - 1;
      }
      scnt = cnt;
    }
    __syncthreads();
    cnt = scnt;
  }
  if (lane == 0) {
    int first = (cnt > 0) ? gidx[0] : 0;
    for (int j = cnt; j < 32; ++j) gidx[j] = first;
  }
  __syncthreads();
  h16* xr = X + ((size_t)b * S + s) * 32 * Kp;
  for (int e = lane; e < 32 * Kp; e += 32) {
    int j = e / Kp, c = e % Kp, g = gidx[j];
    float v = 0.f;
    if (c < 3)            v = xyz[((size_t)b * N + g) * 3 + c] - q[c];
    else if (c < 3 + C)   v = (float)fprev[((size_t)b * N + g) * Cp + (c - 3)];
    xr[(size_t)j * Kp + c] = (h16)v;
  }
}

// ---------------------------------------------------------------------------
// WMMA GEMM: Y[o][m] = sum_k W[o][k] * X[m][k]  (+bias, activation)
// One wave32 per 16x16 output tile, 4 M-tiles per block sharing one O-tile.
// The 16xKp W tile is staged into LDS once per block via the Tensor Data
// Mover (tensor_load_to_lds, TENSORcnt), then A-fragments come from LDS.
// act: 0=relu 1=leaky-relu(0.01) 2=none
// ---------------------------------------------------------------------------
__global__ __launch_bounds__(128) void gemm_wmma(
    const h16* __restrict__ W, const h16* __restrict__ X,
    const float* __restrict__ bias,
    h16* __restrict__ Yh, float* __restrict__ Yf,
    int M, int Kp, int O, int Opad, int act) {
  extern __shared__ h16 wtile[];               // 16 * Kp halves
  int lane = threadIdx.x;                      // 0..31 (wave32)
  int mt = (blockIdx.x * blockDim.y + threadIdx.y) * 16;
  int ot = blockIdx.y * 16;
  int b  = blockIdx.z;

  // ---- TDM stage: copy W[ot..ot+15][0..Kp) -> LDS (wave 0 issues + waits) ----
  if (threadIdx.y == 0) {
    unsigned long long ga = (unsigned long long)(const void*)(W + (size_t)ot * Kp);
    unsigned ldsoff = (unsigned)(unsigned long long)(const void*)wtile;
    int elems = Kp >> 1;                       // 4-byte elements per row
    u32x4 g0;
    g0[0] = 1u;                                // count=1 (user descriptor)
    g0[1] = ldsoff;                            // lds_addr
    g0[2] = (unsigned)(ga & 0xFFFFFFFFull);    // global_addr[31:0]
    g0[3] = (unsigned)((ga >> 32) & 0x1FFFFFFull) | (2u << 30);  // addr[56:32], type=2
    i32x8 g1;
    g1[0] = (2 << 16);                         // data_size=4B, no flags
    g1[1] = (elems & 0xFFFF) << 16;            // tensor_dim0[15:0]
    g1[2] = ((elems >> 16) & 0xFFFF) | (16 << 16);  // tensor_dim0[31:16] | tensor_dim1=16
    g1[3] = (elems & 0xFFFF) << 16;            // tile_dim0 = elems
    g1[4] = 16;                                // tile_dim1 = 16 rows, tile_dim2 = 0
    g1[5] = elems;                             // tensor_dim0_stride[31:0]
    g1[6] = 0;                                 // stride0 hi | dim1_stride lo
    g1[7] = 0;
    i32x4 g2 = {0, 0, 0, 0};                   // 2-D tile: higher dims unused
    i32x4 g3 = {0, 0, 0, 0};
    asm volatile("tensor_load_to_lds %0, %1, %2, %3"
                 :: "s"(g0), "s"(g1), "s"(g2), "s"(g3) : "memory");
    __builtin_amdgcn_s_wait_tensorcnt(0);
  }
  __syncthreads();

  if (mt < M) {
    int r = lane & 15, hh = lane >> 4;
    const h16* wl = wtile + (size_t)r * Kp;                       // A row (LDS)
    const h16* xp = X + ((size_t)b * M + (mt + r)) * Kp + hh * 16; // B column (global)
    v8f acc = {};
    for (int k = 0; k < Kp; k += 32) {
      __builtin_prefetch(xp + k + 128, 0, 1);
      // A fragment: lane(r,h) holds K = {k+h*16..+15}
      v16h a = *(const v16h*)(wl + k + hh * 16);
      v16h bb = *(const v16h*)(xp + k);
      acc = __builtin_amdgcn_wmma_f32_16x16x32_f16(
          /*neg_a=*/false, a, /*neg_b=*/false, bb,
          /*c_mod=*/(short)0, acc, /*reuse_a=*/false, /*reuse_b=*/false);
    }
    // D layout: VGPR v, lanes 0-15 -> row v, lanes 16-31 -> row v+8; col = lane%16
    int m = mt + r;
    int obase = ot + hh * 8;
#pragma unroll
    for (int v = 0; v < 8; ++v) {
      int o = obase + v;
      float val = acc[v];
      if (o < O) {
        val += bias[o];
        if (act == 0)      val = val > 0.f ? val : 0.f;
        else if (act == 1) val = val > 0.f ? val : 0.01f * val;
      } else {
        val = 0.f;
      }
      if (Yh) Yh[((size_t)b * M + m) * Opad + o] = (h16)val;
      if (Yf && o < O) Yf[((size_t)b * M + m) * O + o] = val;
    }
  }
}

// max over 32 samples: Y[b][s*32+j][Op] -> feats[b][s][Op]
__global__ __launch_bounds__(64) void maxpool_ns(const h16* __restrict__ Y,
                                                 h16* __restrict__ feats, int S, int Op) {
  int b = blockIdx.y, s = blockIdx.x;
  const h16* src = Y + ((size_t)b * S + s) * 32 * Op;
  h16* dst = feats + ((size_t)b * S + s) * Op;
  for (int c = threadIdx.x; c < Op; c += 64) {
    float m = -1e30f;
    for (int j = 0; j < 32; ++j) m = fmaxf(m, (float)src[(size_t)j * Op + c]);
    dst[c] = (h16)m;
  }
}

// post conv (170,C) + tanh + global max over S -> codes[b][off+o]
__global__ __launch_bounds__(128) void postconv(const h16* __restrict__ feats,
                                                const float* __restrict__ W,
                                                const float* __restrict__ bias,
                                                float* __restrict__ codes,
                                                int S, int C, int off) {
  __shared__ float red[128];
  int b = blockIdx.y, o = blockIdx.x, t = threadIdx.x;
  const float* wr = W + (size_t)o * C;
  float lm = -1e30f;
  for (int s = t; s < S; s += 128) {
    const h16* f = feats + ((size_t)b * S + s) * C;
    float acc = bias[o];
    for (int c = 0; c < C; ++c) acc += (float)f[c] * wr[c];
    lm = fmaxf(lm, tanhf(acc));
  }
  red[t] = lm;
  __syncthreads();
  for (int w = 64; w > 0; w >>= 1) {
    if (t < w) red[t] = fmaxf(red[t], red[t + w]);
    __syncthreads();
  }
  if (t == 0) codes[(size_t)b * 510 + off + o] = red[0];
}

// f32 [M][K] -> f16 [M][Kp] zero padded
__global__ void pack_rows(const float* __restrict__ src, h16* __restrict__ dst,
                          int M, int K, int Kp) {
  int i = blockIdx.x * blockDim.x + threadIdx.x;
  if (i >= M * Kp) return;
  int m = i / Kp, k = i % Kp;
  dst[i] = (k < K) ? (h16)src[(size_t)m * K + k] : (h16)0.f;
}

// concat(a,b) -> f16 [M][Kp], cols [0,K)=a, [K,2K)=b, rest 0
__global__ void pack_concat2(const float* __restrict__ a, const float* __restrict__ b,
                             h16* __restrict__ dst, int M, int K, int Kp) {
  int i = blockIdx.x * blockDim.x + threadIdx.x;
  if (i >= M * Kp) return;
  int m = i / Kp, k = i % Kp;
  float v = 0.f;
  if (k < K)          v = a[(size_t)m * K + k];
  else if (k < 2 * K) v = b[(size_t)m * K + (k - K)];
  dst[i] = (h16)v;
}

__global__ void cage_out(const float* __restrict__ y, const float* __restrict__ tmpl,
                         float* __restrict__ out) {
  int i = blockIdx.x * blockDim.x + threadIdx.x;
  if (i >= BATCH * 3 * CAGEP) return;
  out[i] = y[i] + tmpl[i % (3 * CAGEP)];
}

__global__ void newcage_out(const float* __restrict__ y, const float* __restrict__ cage,
                            float* __restrict__ out) {
  int i = blockIdx.x * blockDim.x + threadIdx.x;
  if (i >= BATCH * 3 * CAGEP) return;
  out[i] = y[i] + cage[i];
}

// ---------------------------------------------------------------------------
// Host orchestration
// ---------------------------------------------------------------------------
extern "C" void kernel_launch(void* const* d_in, const int* in_sizes, int n_in,
                              void* d_out, int out_size, void* d_ws, size_t ws_size,
                              hipStream_t stream) {
  (void)n_in; (void)out_size; (void)ws_size;

  // ---- map inputs (handles both insertion-order and sorted-jax-tree order) ----
  const float *src, *tgt, *tmpl, *mw, *mb;
  const float *sa_w[3][3], *sa_b[3][3], *post_w[3], *post_b[3];
  const float *nc_w[3], *nc_b[3], *nd_w[3], *nd_b[3];
  auto IN = [&](int i) { return (const float*)d_in[i]; };
  bool insertion = (in_sizes[0] == BATCH * 3 * NPTS);
  if (insertion) {
    src = IN(0); tgt = IN(1);
    int k = 2;
    for (int l = 0; l < 3; ++l) {
      for (int j = 0; j < 3; ++j) sa_w[l][j] = IN(k++);
      for (int j = 0; j < 3; ++j) sa_b[l][j] = IN(k++);
    }
    for (int l = 0; l < 3; ++l) { post_w[l] = IN(k++); post_b[l] = IN(k++); }
    for (int j = 0; j < 3; ++j) { nc_w[j] = IN(k++); nc_b[j] = IN(k++); }
    for (int j = 0; j < 3; ++j) { nd_w[j] = IN(k++); nd_b[j] = IN(k++); }
    mw = IN(k++); mb = IN(k++); tmpl = IN(k++);
  } else {  // sorted keys: merger,nc,nd,post,sa,template then source,target
    mw = IN(0); mb = IN(1);
    int k = 2;
    for (int j = 0; j < 3; ++j) { nc_w[j] = IN(k++); nc_b[j] = IN(k++); }
    for (int j = 0; j < 3; ++j) { nd_w[j] = IN(k++); nd_b[j] = IN(k++); }
    for (int l = 0; l < 3; ++l) { post_b[l] = IN(k++); post_w[l] = IN(k++); }
    for (int l = 0; l < 3; ++l) {
      for (int j = 0; j < 3; ++j) sa_b[l][j] = IN(k++);
      for (int j = 0; j < 3; ++j) sa_w[l][j] = IN(k++);
    }
    tmpl = IN(k++); src = IN(k++); tgt = IN(k++);
  }

  // ---- workspace bump allocator ----
  char* wp = (char*)d_ws;
  auto alloc = [&](size_t bytes) -> void* {
    void* r = (void*)wp;
    wp += (bytes + 255) & ~(size_t)255;
    return r;
  };
  float* xyz0  = (float*)alloc((size_t)BATCH * NPTS * 3 * 4);
  float* nxyz0 = (float*)alloc((size_t)BATCH * 512 * 3 * 4);
  float* nxyz1 = (float*)alloc((size_t)BATCH * 128 * 3 * 4);
  float* nxyz2 = (float*)alloc((size_t)BATCH * 32 * 3 * 4);
  int*   idx0  = (int*)alloc((size_t)BATCH * 512 * 4);
  int*   idx1  = (int*)alloc((size_t)BATCH * 128 * 4);
  int*   idx2  = (int*)alloc((size_t)BATCH * 32 * 4);
  float* codeS = (float*)alloc((size_t)BATCH * 510 * 4);
  float* codeT = (float*)alloc((size_t)BATCH * 510 * 4);
  float* ync   = (float*)alloc((size_t)BATCH * 126 * 4);
  float* ynd   = (float*)alloc((size_t)BATCH * 126 * 4);

  h16* Xb = (h16*)alloc((size_t)8388608 * 2);
  h16* Y0 = (h16*)alloc((size_t)8388608 * 2);
  h16* Y1 = (h16*)alloc((size_t)16777216 * 2);
  h16* Y2 = (h16*)alloc((size_t)25165824 * 2);
  h16* feats[3];
  feats[0] = (h16*)alloc((size_t)BATCH * 512 * 96 * 2);
  feats[1] = (h16*)alloc((size_t)BATCH * 128 * 192 * 2);
  feats[2] = (h16*)alloc((size_t)BATCH * 32 * 256 * 2);
  h16* xpack = (h16*)alloc((size_t)16 * 1024 * 2);
  h16* mout  = (h16*)alloc((size_t)16 * 1024 * 2);
  h16* dy1   = (h16*)alloc((size_t)16 * 512 * 2);
  h16* dy2   = (h16*)alloc((size_t)16 * 256 * 2);

  // fp16 weight buffers
  static const int OlT[3][3]  = {{24, 48, 96}, {48, 96, 192}, {96, 192, 256}};
  static const int OpT[3][3]  = {{32, 64, 96}, {64, 96, 192}, {96, 192, 256}};
  static const int KsT[3][3]  = {{3, 24, 48}, {99, 48, 96}, {195, 96, 192}};
  static const int KpX[3]     = {32, 128, 224};
  h16* saWd[3][3];
  for (int l = 0; l < 3; ++l)
    for (int j = 0; j < 3; ++j) {
      int Kp = (j == 0) ? KpX[l] : OpT[l][j - 1];
      saWd[l][j] = (h16*)alloc((size_t)OpT[l][j] * Kp * 2);
    }
  h16* ncWd[3] = {(h16*)alloc((size_t)512 * 512 * 2),
                  (h16*)alloc((size_t)256 * 512 * 2),
                  (h16*)alloc((size_t)128 * 256 * 2)};
  h16* ndWd[3] = {(h16*)alloc((size_t)512 * 1024 * 2),
                  (h16*)alloc((size_t)256 * 512 * 2),
                  (h16*)alloc((size_t)128 * 256 * 2)};
  h16* mWd = (h16*)alloc((size_t)1024 * 1024 * 2);

  // ---- convert weights to fp16 tiles ----
  auto cvt = [&](const float* s, h16* d, int O, int K, int Op, int Kp) {
    int tot = Op * Kp;
    convert_w<<<(tot + 255) / 256, 256, 0, stream>>>(s, d, O, K, Op, Kp);
  };
  for (int l = 0; l < 3; ++l)
    for (int j = 0; j < 3; ++j) {
      int Kp = (j == 0) ? KpX[l] : OpT[l][j - 1];
      cvt(sa_w[l][j], saWd[l][j], OlT[l][j], KsT[l][j], OpT[l][j], Kp);
    }
  cvt(nc_w[0], ncWd[0], 512, 510, 512, 512);
  cvt(nc_w[1], ncWd[1], 256, 512, 256, 512);
  cvt(nc_w[2], ncWd[2], 126, 256, 128, 256);
  cvt(nd_w[0], ndWd[0], 512, 1020, 512, 1024);
  cvt(nd_w[1], ndWd[1], 256, 512, 256, 512);
  cvt(nd_w[2], ndWd[2], 126, 256, 128, 256);
  cvt(mw, mWd, 1020, 1020, 1024, 1024);

  auto gemm = [&](const h16* W, const h16* X, const float* bias, h16* Yh, float* Yf,
                  int M, int Kp, int O, int Op, int act, int nb) {
    dim3 grid((M / 16 + 3) / 4, Op / 16, nb);
    dim3 block(32, 4, 1);
    size_t shmem = (size_t)Kp * 32;  // 16 rows * Kp halves * 2B
    gemm_wmma<<<grid, block, shmem, stream>>>(W, X, bias, Yh, Yf, M, Kp, O, Op, act);
  };

  // ---- two encodes ----
  const float* shapes[2] = {src, tgt};
  float* codebuf[2] = {codeS, codeT};
  const int Ns[3] = {NPTS, 512, 128};
  const int Ss[3] = {512, 128, 32};
  const float R2[3] = {0.04f, 0.16f, 0.64f};
  const int Cprev[3] = {0, 96, 192};

  for (int si = 0; si < 2; ++si) {
    transpose_pc<<<(BATCH * NPTS * 3 + 255) / 256, 256, 0, stream>>>(shapes[si], xyz0);
    const float* xyzs[3] = {xyz0, nxyz0, nxyz1};
    float* nxs[3] = {nxyz0, nxyz1, nxyz2};
    int* idxs[3] = {idx0, idx1, idx2};
    const h16* fprev[3] = {nullptr, feats[0], feats[1]};
    for (int l = 0; l < 3; ++l) {
      int N = Ns[l], S = Ss[l], M = S * 32;
      fps_kernel<<<BATCH, 256, 0, stream>>>(xyzs[l], idxs[l], N, S);
      int tg = BATCH * S * 3;
      gather_xyz<<<(tg + 255) / 256, 256, 0, stream>>>(xyzs[l], idxs[l], nxs[l], N, S);
      ballgroup<<<dim3(S, BATCH), 32, 0, stream>>>(xyzs[l], nxs[l], fprev[l], Xb,
                                                   N, S, Cprev[l], Cprev[l], KpX[l], R2[l]);
      const h16* xin = Xb;
      int kp = KpX[l];
      h16* youts[3] = {Y0, Y1, Y2};
      for (int j = 0; j < 3; ++j) {
        gemm(saWd[l][j], xin, sa_b[l][j], youts[j], nullptr,
             M, kp, OlT[l][j], OpT[l][j], /*relu*/ 0, BATCH);
        xin = youts[j];
        kp = OpT[l][j];
      }
      maxpool_ns<<<dim3(S, BATCH), 64, 0, stream>>>(Y2, feats[l], S, OpT[l][2]);
      postconv<<<dim3(170, BATCH), 128, 0, stream>>>(feats[l], post_w[l], post_b[l],
                                                     codebuf[si], S, OpT[l][2], 170 * l);
    }
  }

  // ---- cage deformer (nc) on source code ----
  pack_rows<<<(16 * 512 + 255) / 256, 256, 0, stream>>>(codeS, xpack, 16, 510, 512);
  gemm(ncWd[0], xpack, nc_b[0], dy1, nullptr, 16, 512, 512, 512, /*lrelu*/ 1, 1);
  gemm(ncWd[1], dy1, nc_b[1], dy2, nullptr, 16, 512, 256, 256, 1, 1);
  gemm(ncWd[2], dy2, nc_b[2], nullptr, ync, 16, 256, 126, 128, /*none*/ 2, 1);
  float* outCage = (float*)d_out;
  float* outNew  = (float*)d_out + BATCH * 3 * CAGEP;
  cage_out<<<(BATCH * 126 + 255) / 256, 256, 0, stream>>>(ync, tmpl, outCage);

  // ---- merger + target deformer (nd) ----
  pack_concat2<<<(16 * 1024 + 255) / 256, 256, 0, stream>>>(codeS, codeT, xpack, 16, 510, 1024);
  gemm(mWd, xpack, mb, mout, nullptr, 16, 1024, 1020, 1024, 1, 1);
  gemm(ndWd[0], mout, nd_b[0], dy1, nullptr, 16, 1024, 512, 512, 1, 1);
  gemm(ndWd[1], dy1, nd_b[1], dy2, nullptr, 16, 512, 256, 256, 1, 1);
  gemm(ndWd[2], dy2, nd_b[2], nullptr, ynd, 16, 256, 126, 128, 2, 1);
  newcage_out<<<(BATCH * 126 + 255) / 256, 256, 0, stream>>>(ynd, outCage, outNew);
}
